// VDEncoder_66709432041700
// MI455X (gfx1250) — compile-verified, matrix operand order checked by reading, not run on hardware
//
#include <hip/hip_runtime.h>

// ---------------------------------------------------------------------------
// Fused 3-layer LSTM + locked dropout for MI455X (gfx1250, wave32, WMMA).
// One wave32 per 16-row batch tile runs all T=1024 steps of all 3 layers.
// Gate GEMMs use v_wmma_f32_16x16x32_f16; h-state relayout via LDS.
// ---------------------------------------------------------------------------

typedef _Float16 v16h __attribute__((ext_vector_type(16)));
typedef _Float16 v8h  __attribute__((ext_vector_type(8)));
typedef float    v8f  __attribute__((ext_vector_type(8)));
typedef float    v4f  __attribute__((ext_vector_type(4)));

#define WMMA_F16F32(a, b, c) \
  __builtin_amdgcn_wmma_f32_16x16x32_f16(false, (a), false, (b), (short)0, (c), false, false)

__device__ __forceinline__ float frcp(float x) {
#if __has_builtin(__builtin_amdgcn_rcpf)
  return __builtin_amdgcn_rcpf(x);
#else
  return 1.0f / x;
#endif
}
__device__ __forceinline__ float fsig(float x) { return frcp(1.0f + __expf(-x)); }
__device__ __forceinline__ float ftanh_(float x) {
#if __has_builtin(__builtin_amdgcn_tanhf)
  return __builtin_amdgcn_tanhf(x);
#else
  return 2.0f * fsig(2.0f * x) - 1.0f;
#endif
}

// Load a 16x32 f16 A/B fragment (row-major, 32-half row stride) per the CDNA5
// 16-bit 16x32 layout: lane<16 -> row=lane, K {0..7,16..23}; lane>=16 -> row=lane-16,
// K {8..15,24..31}. Two ds_load_b128 per lane.
__device__ __forceinline__ v16h load_frag(const _Float16* base, int lane) {
  const int row = lane & 15;
  const int ko  = (lane >> 4) << 3;   // 0 or 8
  const _Float16* p = base + row * 32 + ko;
  v8h lo = *(const v8h*)(p);          // K = ko .. ko+7
  v8h hi = *(const v8h*)(p + 16);     // K = ko+16 .. ko+23
  v16h r;
#pragma unroll
  for (int i = 0; i < 8; ++i) { r[i] = lo[i]; r[i + 8] = hi[i]; }
  return r;
}

// Store a 16x16 f32 C-tile (v8f per lane) into a row-major 16x32 f16 buffer at
// column offset 16*k. C layout: VGPR r, lane l -> (row = r + 8*(l>=16), col = (l&15)).
__device__ __forceinline__ void store_tile_f16(_Float16* buf, v8f c, int k, int lane) {
  const int colp = (lane & 15) + (k << 4);
  const int rb   = (lane >> 4) << 3;
#pragma unroll
  for (int r = 0; r < 8; ++r) buf[(rb + r) * 32 + colp] = (_Float16)c[r];
}

__global__ __launch_bounds__(32)
void lstm3_fused_wmma(const float* __restrict__ x,
                      const float* __restrict__ Wih1, const float* __restrict__ Whh1,
                      const float* __restrict__ bih1, const float* __restrict__ bhh1,
                      const float* __restrict__ Wih2, const float* __restrict__ Whh2,
                      const float* __restrict__ bih2, const float* __restrict__ bhh2,
                      const float* __restrict__ Wih3, const float* __restrict__ Whh3,
                      const float* __restrict__ bih3, const float* __restrict__ bhh3,
                      const float* __restrict__ mask1, const float* __restrict__ mask2,
                      const float* __restrict__ mask3,
                      float* __restrict__ out) {
  // ---- LDS ----
  __shared__ _Float16 lWih1[128 * 32];   // layer1 input weights, f16, gate-row-major (N x K)
  __shared__ _Float16 lWhh1[128 * 32];   // layer1 recurrent weights
  __shared__ _Float16 lWih2[32 * 32];    // layer2 input weights (K=32)
  __shared__ _Float16 lWhh2[32 * 32];    // layer2 recurrent (K=8, zero-padded to 32)
  __shared__ _Float16 lW3[32 * 32];      // layer3 [Wih3 | Whh3] (K=16, zero-padded to 32)
  __shared__ float    lb1[128], lb2[32], lb3[32];
  __shared__ _Float16 hbuf1[16 * 32];    // h1 (f16) for next-step recurrence A-frag
  __shared__ _Float16 h1mbuf[16 * 32];   // h1 * mask1: layer2 input A-frag
  __shared__ _Float16 hbuf2[16 * 32];    // h2 in cols 0..7, zeros elsewhere
  __shared__ _Float16 buf3[16 * 32];     // h2m in cols 0..7, h3 in cols 8..15, zeros rest
  __shared__ float    gbuf[16 * 32];     // f32 gate staging for layers 2/3 elementwise

  const int lane = threadIdx.x;          // 0..31, single wave per block
  const int b0   = blockIdx.x * 16;      // batch tile base
  const int cl   = lane & 15;            // C-tile column (within 16)
  const int rb   = (lane >> 4) << 3;     // C-tile row base (0 or 8)

  // ---- prologue: weights -> f16 LDS ----
  for (int i = lane; i < 128 * 32; i += 32) {
    lWih1[i] = (_Float16)Wih1[i];
    lWhh1[i] = (_Float16)Whh1[i];
  }
  for (int i = lane; i < 32 * 32; i += 32) {
    const int r = i >> 5, c = i & 31;
    lWih2[i] = (_Float16)Wih2[i];
    lWhh2[i] = (c < 8) ? (_Float16)Whh2[r * 8 + c] : (_Float16)0.0f;
    _Float16 v = (_Float16)0.0f;
    if (c < 8)       v = (_Float16)Wih3[r * 8 + c];
    else if (c < 16) v = (_Float16)Whh3[r * 8 + (c - 8)];
    lW3[i] = v;
  }
  for (int i = lane; i < 128; i += 32) lb1[i] = bih1[i] + bhh1[i];
  if (lane < 32) { lb2[lane] = bih2[lane] + bhh2[lane]; lb3[lane] = bih3[lane] + bhh3[lane]; }
  for (int i = lane; i < 16 * 32; i += 32) {
    hbuf1[i] = (_Float16)0.0f; h1mbuf[i] = (_Float16)0.0f;
    hbuf2[i] = (_Float16)0.0f; buf3[i] = (_Float16)0.0f;
  }
  __syncthreads();

  // ---- layer 2/3 weight fragments held in registers for all 1024 steps ----
  v16h fWih2[2], fWhh2[2], fW3[2];
#pragma unroll
  for (int n = 0; n < 2; ++n) {
    fWih2[n] = load_frag(lWih2 + n * 16 * 32, lane);
    fWhh2[n] = load_frag(lWhh2 + n * 16 * 32, lane);
    fW3[n]   = load_frag(lW3   + n * 16 * 32, lane);
  }

  // ---- locked-dropout masks (constant over t) ----
  v8f m1[2];
#pragma unroll
  for (int k = 0; k < 2; ++k)
#pragma unroll
    for (int r = 0; r < 8; ++r)
      m1[k][r] = mask1[(size_t)(b0 + rb + r) * 32 + cl + 16 * k];
  float m2r[8], m3r[8];
  {
    const int row = lane & 15;
    const v4f* m2p = (const v4f*)(mask2 + (size_t)(b0 + row) * 8);
    const v4f* m3p = (const v4f*)(mask3 + (size_t)(b0 + row) * 8);
    v4f a = m2p[0], b = m2p[1], c = m3p[0], d = m3p[1];
#pragma unroll
    for (int j = 0; j < 4; ++j) { m2r[j] = a[j]; m2r[4 + j] = b[j]; m3r[j] = c[j]; m3r[4 + j] = d[j]; }
  }

  // ---- state ----
  v8f c1[2];
#pragma unroll
  for (int k = 0; k < 2; ++k)
#pragma unroll
    for (int r = 0; r < 8; ++r) c1[k][r] = 0.0f;
  float c2s[8], c3s[8];
#pragma unroll
  for (int j = 0; j < 8; ++j) { c2s[j] = 0.0f; c3s[j] = 0.0f; }

  // ======================= time loop =======================
  for (int t = 0; t < 1024; ++t) {
    // ---- x A-fragment straight from global (f32 -> f16) ----
    const float* xp = x + ((size_t)t * 512 + b0 + cl) * 32 + rb;  // rb doubles as K offset (0/8)
    v16h ax;
    {
      v4f f0 = *(const v4f*)(xp);
      v4f f1 = *(const v4f*)(xp + 4);
      v4f f2 = *(const v4f*)(xp + 16);
      v4f f3 = *(const v4f*)(xp + 20);
#pragma unroll
      for (int i = 0; i < 4; ++i) {
        ax[i] = (_Float16)f0[i];  ax[4 + i]  = (_Float16)f1[i];
        ax[8 + i] = (_Float16)f2[i]; ax[12 + i] = (_Float16)f3[i];
      }
    }
    v16h ah1 = load_frag(hbuf1, lane);

    // ---- layer 1 gates: 8 N-tiles x (x-proj + h-proj) WMMAs, B streamed from LDS ----
    v8f g[8];
#pragma unroll
    for (int n = 0; n < 8; ++n) {
      const float bv = lb1[n * 16 + cl];
      v8f c;
#pragma unroll
      for (int r = 0; r < 8; ++r) c[r] = bv;
      v16h bA = load_frag(lWih1 + n * 16 * 32, lane);
      c = WMMA_F16F32(ax, bA, c);
      v16h bB = load_frag(lWhh1 + n * 16 * 32, lane);
      c = WMMA_F16F32(ah1, bB, c);
      g[n] = c;
    }

    // ---- layer 1 elementwise (gate order i,f,g,o; each gate = 2 tiles) ----
#pragma unroll
    for (int k = 0; k < 2; ++k) {
      v8f h1t, h1mt;
#pragma unroll
      for (int r = 0; r < 8; ++r) {
        const float iv = fsig(g[0 + k][r]);
        const float fv = fsig(g[2 + k][r]);
        const float gv = ftanh_(g[4 + k][r]);
        const float ov = fsig(g[6 + k][r]);
        const float cc = fv * c1[k][r] + iv * gv;
        c1[k][r] = cc;
        const float hv = ov * ftanh_(cc);
        h1t[r]  = hv;
        h1mt[r] = hv * m1[k][r];
      }
      store_tile_f16(hbuf1, h1t, k, lane);    // unmasked h1 for recurrence
      store_tile_f16(h1mbuf, h1mt, k, lane);  // masked h1: layer2 input
    }
    __syncthreads();

    // ---- layer 2: g2 = h1m @ Wih2^T + h2 @ Whh2^T + b2 (2 N-tiles) ----
    v16h a2  = load_frag(h1mbuf, lane);
    v16h ah2 = load_frag(hbuf2, lane);
#pragma unroll
    for (int n = 0; n < 2; ++n) {
      const float bv = lb2[n * 16 + cl];
      v8f c;
#pragma unroll
      for (int r = 0; r < 8; ++r) c[r] = bv;
      c = WMMA_F16F32(a2, fWih2[n], c);
      c = WMMA_F16F32(ah2, fWhh2[n], c);
#pragma unroll
      for (int r = 0; r < 8; ++r) gbuf[(rb + r) * 32 + cl + 16 * n] = c[r];
    }
    __syncthreads();

    if (lane < 16) {
      const int row = lane;
      const float* gr = gbuf + row * 32;
#pragma unroll
      for (int j = 0; j < 8; ++j) {
        const float iv = fsig(gr[j]);
        const float fv = fsig(gr[8 + j]);
        const float gv = ftanh_(gr[16 + j]);
        const float ov = fsig(gr[24 + j]);
        const float cc = fv * c2s[j] + iv * gv;
        c2s[j] = cc;
        const float hv = ov * ftanh_(cc);
        hbuf2[row * 32 + j] = (_Float16)hv;            // h2 state (recurrence)
        buf3[row * 32 + j]  = (_Float16)(hv * m2r[j]); // masked h2: layer3 input
      }
    }
    __syncthreads();

    // ---- layer 3: A = [h2m | h3 | 0] (K=16 padded), single combined B ----
    v16h a3 = load_frag(buf3, lane);
#pragma unroll
    for (int n = 0; n < 2; ++n) {
      const float bv = lb3[n * 16 + cl];
      v8f c;
#pragma unroll
      for (int r = 0; r < 8; ++r) c[r] = bv;
      c = WMMA_F16F32(a3, fW3[n], c);
#pragma unroll
      for (int r = 0; r < 8; ++r) gbuf[(rb + r) * 32 + cl + 16 * n] = c[r];
    }
    __syncthreads();

    if (lane < 16) {
      const int row = lane;
      const float* gr = gbuf + row * 32;
      v4f o0, o1;
#pragma unroll
      for (int j = 0; j < 8; ++j) {
        const float iv = fsig(gr[j]);
        const float fv = fsig(gr[8 + j]);
        const float gv = ftanh_(gr[16 + j]);
        const float ov = fsig(gr[24 + j]);
        const float cc = fv * c3s[j] + iv * gv;
        c3s[j] = cc;
        const float hv = ov * ftanh_(cc);
        buf3[row * 32 + 8 + j] = (_Float16)hv;   // h3 state for next step
        const float hm = hv * m3r[j];
        if (j < 4) o0[j] = hm; else o1[j - 4] = hm;
      }
      float* op = out + ((size_t)t * 512 + b0 + row) * 8;
      *(v4f*)(op)     = o0;
      *(v4f*)(op + 4) = o1;
    }
    __syncthreads();
  }
}

extern "C" void kernel_launch(void* const* d_in, const int* in_sizes, int n_in,
                              void* d_out, int out_size, void* d_ws, size_t ws_size,
                              hipStream_t stream) {
  (void)in_sizes; (void)n_in; (void)out_size; (void)d_ws; (void)ws_size;
  const float* x     = (const float*)d_in[0];
  const float* Wih1  = (const float*)d_in[1];
  const float* Whh1  = (const float*)d_in[2];
  const float* bih1  = (const float*)d_in[3];
  const float* bhh1  = (const float*)d_in[4];
  const float* Wih2  = (const float*)d_in[5];
  const float* Whh2  = (const float*)d_in[6];
  const float* bih2  = (const float*)d_in[7];
  const float* bhh2  = (const float*)d_in[8];
  const float* Wih3  = (const float*)d_in[9];
  const float* Whh3  = (const float*)d_in[10];
  const float* bih3  = (const float*)d_in[11];
  const float* bhh3  = (const float*)d_in[12];
  const float* mask1 = (const float*)d_in[13];
  const float* mask2 = (const float*)d_in[14];
  const float* mask3 = (const float*)d_in[15];
  float* out = (float*)d_out;

  // 512 batch rows / 16 per tile = 32 independent wave32 workgroups.
  lstm3_fused_wmma<<<dim3(32), dim3(32), 0, stream>>>(
      x, Wih1, Whh1, bih1, bhh1, Wih2, Whh2, bih2, bhh2,
      Wih3, Whh3, bih3, bhh3, mask1, mask2, mask3, out);
}
